// GraphNeuralNetwork_25048249270891
// MI455X (gfx1250) — matrix operand order repaired and checked by reading, not compiled, not run on hardware
//
#include <hip/hip_runtime.h>
#include <hip/hip_bf16.h>
#include <math.h>

typedef __attribute__((ext_vector_type(16))) _Float16 v16h;
typedef __attribute__((ext_vector_type(8)))  float    v8f;

#define NEG_SLOPE 0.2f

// -------------------- helpers --------------------

__device__ __forceinline__ float atomic_max_f32(float* addr, float val) {
  // sign-split trick: int ordering == float ordering for x>=0,
  // reversed-unsigned ordering for x<0. emax initialized to -INF.
  if (val >= 0.0f)
    return __int_as_float(atomicMax((int*)addr, __float_as_int(val)));
  else
    return __uint_as_float(atomicMin((unsigned int*)addr, __float_as_uint(val)));
}

// -------------------- kernels --------------------

// WT[o][i] = (f16) W[i][o]   (W is [din, dout] row-major)
__global__ void k_convert_wt(const float* __restrict__ W, _Float16* __restrict__ WT,
                             int din, int dout) {
  int gid = blockIdx.x * blockDim.x + threadIdx.x;
  int total = din * dout;
  if (gid >= total) return;
  int o = gid / din, i = gid - o * din;
  WT[(size_t)o * din + i] = (_Float16)W[(size_t)i * dout + o];
}

// h[n][o] = relu(x[n][:16] @ Wp[:,o] + bp[o])
__global__ void k_proj(const float* __restrict__ x, const float* __restrict__ Wp,
                       const float* __restrict__ bp, float* __restrict__ h,
                       int n_nodes) {
  int gid = blockIdx.x * blockDim.x + threadIdx.x;
  if (gid >= n_nodes * 64) return;
  int n = gid >> 6, o = gid & 63;
  const float* xr = x + (size_t)n * 16;
  float s = bp[o];
#pragma unroll
  for (int k = 0; k < 16; ++k) s += xr[k] * Wp[k * 64 + o];
  h[(size_t)n * 64 + o] = fmaxf(s, 0.0f);
}

// C[n_rows, dout] = A[n_rows, DIN] (f32 -> f16) @ BT^T using v_wmma_f32_16x16x32_f16.
// BT is [dout][DIN] f16 (pre-transposed weights).
// Each wave computes a 16x32 strip of C (two 16x16 n-tiles sharing one A fragment);
// 4 waves per block cover 128 output columns. DIN is compile-time -> K-loop unrolls.
template <int DIN>
__global__ void __launch_bounds__(128)
k_gemm_wmma(const float* __restrict__ A, const _Float16* __restrict__ BT,
            float* __restrict__ C, int n_rows, int dout) {
  const int lane  = threadIdx.x & 31;
  const int wave  = threadIdx.x >> 5;
  const int mrow  = lane & 15;   // A: M row / B,D: N column
  const int khalf = lane >> 4;   // which K-half this lane holds
  const int m_tile  = blockIdx.x;
  const int colbase = blockIdx.y * 128 + wave * 32;   // dout % 32 == 0 always
  if (colbase >= dout) return;

  int arow = m_tile * 16 + mrow;
  if (arow >= n_rows) arow = n_rows - 1;  // clamp reads; stores guarded below
  const float*    Ap  = A  + (size_t)arow * DIN;
  const _Float16* Bp0 = BT + (size_t)(colbase + mrow) * DIN;
  const _Float16* Bp1 = BT + (size_t)(colbase + 16 + mrow) * DIN;

  v8f acc0 = {}, acc1 = {};
#pragma unroll
  for (int k0 = 0; k0 < DIN; k0 += 32) {
    v16h a, b0, b1;
#pragma unroll
    for (int g = 0; g < 8; ++g) {
      // A 16x32 f16: VGPR g<4 -> K = khalf*8 + 2g ; g>=4 -> K = 16 + khalf*8 + 2(g-4)
      int ka = (g < 4) ? (khalf * 8 + 2 * g) : (16 + khalf * 8 + 2 * (g - 4));
      a[2 * g]     = (_Float16)Ap[k0 + ka];
      a[2 * g + 1] = (_Float16)Ap[k0 + ka + 1];
      // B 32x16 f16: same K mapping as A
      int kb = (g < 4) ? (khalf * 8 + 2 * g) : (16 + khalf * 8 + 2 * (g - 4));
      b0[2 * g]     = Bp0[k0 + kb];
      b0[2 * g + 1] = Bp0[k0 + kb + 1];
      b1[2 * g]     = Bp1[k0 + kb];
      b1[2 * g + 1] = Bp1[k0 + kb + 1];
    }
    acc0 = __builtin_amdgcn_wmma_f32_16x16x32_f16(false, a, false, b0, (short)0,
                                                  acc0, false, false);
    acc1 = __builtin_amdgcn_wmma_f32_16x16x32_f16(false, a, false, b1, (short)0,
                                                  acc1, false, false);
  }

  // D layout: VGPR r -> M = r + 8*khalf, N = lane&15
  float* Crow = C + (size_t)(m_tile * 16 + 8 * khalf) * dout + colbase + mrow;
  if (m_tile * 16 + 16 <= n_rows) {
    // full tile: branch-free stores
#pragma unroll
    for (int r = 0; r < 8; ++r) {
      Crow[(size_t)r * dout]      = acc0[r];
      Crow[(size_t)r * dout + 16] = acc1[r];
    }
  } else {
#pragma unroll
    for (int r = 0; r < 8; ++r) {
      int m = m_tile * 16 + r + 8 * khalf;
      if (m < n_rows) {
        Crow[(size_t)r * dout]      = acc0[r];
        Crow[(size_t)r * dout + 16] = acc1[r];
      }
    }
  }
}

// alpha_src/dst[n][h] = <hproj[n,h,:], a_src/dst[h,:]> ; also init emax=-inf, denom=0
__global__ void k_alpha_init(const float* __restrict__ hp, const float* __restrict__ avs,
                             const float* __restrict__ avd, float* __restrict__ asrc,
                             float* __restrict__ adst, float* __restrict__ emax,
                             float* __restrict__ denom, int n_nodes, int H) {
  int gid = blockIdx.x * blockDim.x + threadIdx.x;
  if (gid >= n_nodes * H) return;
  int n = gid / H, h = gid - n * H;
  const float* row = hp + (size_t)n * (H * 64) + h * 64;
  const float* vs  = avs + h * 64;
  const float* vd  = avd + h * 64;
  float ss = 0.f, sd = 0.f;
#pragma unroll 8
  for (int c = 0; c < 64; ++c) { float v = row[c]; ss += v * vs[c]; sd += v * vd[c]; }
  asrc[gid] = ss;
  adst[gid] = sd;
  emax[gid]  = -INFINITY;
  denom[gid] = 0.0f;
}

__global__ void k_zero(float* __restrict__ p, long long n) {
  long long i = (long long)blockIdx.x * blockDim.x + threadIdx.x;
  if (i < n) p[i] = 0.0f;
}

__device__ __forceinline__ void edge_sd(int e, int n_edges,
                                        const int* __restrict__ srcs,
                                        const int* __restrict__ dsts, int& s, int& d) {
  if (e < n_edges) { s = srcs[e]; d = dsts[e]; }
  else             { s = d = e - n_edges; }   // appended self loops
}

// pass 1: segment max of leaky_relu(asrc[s]+adst[d]) over incoming edges of d
__global__ void k_edge_max(const int* __restrict__ srcs, const int* __restrict__ dsts,
                           int n_edges, int n_nodes, const float* __restrict__ asrc,
                           const float* __restrict__ adst, float* __restrict__ emax,
                           int H) {
  int e = blockIdx.x * blockDim.x + threadIdx.x;
  if (e >= n_edges + n_nodes) return;
  int s, d; edge_sd(e, n_edges, srcs, dsts, s, d);
  for (int h = 0; h < H; ++h) {
    float v = asrc[s * H + h] + adst[d * H + h];
    v = v > 0.0f ? v : NEG_SLOPE * v;
    atomic_max_f32(&emax[d * H + h], v);
  }
}

// pass 2: denom[d] = sum exp(e - emax[d])
__global__ void k_edge_sum(const int* __restrict__ srcs, const int* __restrict__ dsts,
                           int n_edges, int n_nodes, const float* __restrict__ asrc,
                           const float* __restrict__ adst, const float* __restrict__ emax,
                           float* __restrict__ denom, int H) {
  int e = blockIdx.x * blockDim.x + threadIdx.x;
  if (e >= n_edges + n_nodes) return;
  int s, d; edge_sd(e, n_edges, srcs, dsts, s, d);
  for (int h = 0; h < H; ++h) {
    float v = asrc[s * H + h] + adst[d * H + h];
    v = v > 0.0f ? v : NEG_SLOPE * v;
    atomicAdd(&denom[d * H + h], expf(v - emax[d * H + h]));
  }
}

// pass 3: out[d, h, c] += alpha(e,h) * hproj[s, h, c]   (thread per (edge, c))
__global__ void k_edge_msg(const int* __restrict__ srcs, const int* __restrict__ dsts,
                           int n_edges, int n_nodes, const float* __restrict__ asrc,
                           const float* __restrict__ adst, const float* __restrict__ emax,
                           const float* __restrict__ denom, const float* __restrict__ hp,
                           float* __restrict__ out, int H) {
  long long t = (long long)blockIdx.x * blockDim.x + threadIdx.x;
  long long tot = (long long)(n_edges + n_nodes) * 64;
  if (t >= tot) return;
  int e = (int)(t >> 6);
  int c = (int)(t & 63);
  int s, d; edge_sd(e, n_edges, srcs, dsts, s, d);
  int D = H * 64;
  for (int h = 0; h < H; ++h) {
    float v = asrc[s * H + h] + adst[d * H + h];
    v = v > 0.0f ? v : NEG_SLOPE * v;
    float alpha = expf(v - emax[d * H + h]) / denom[d * H + h];
    atomicAdd(&out[(size_t)d * D + h * 64 + c],
              hp[(size_t)s * D + h * 64 + c] * alpha);
  }
}

__global__ void k_bias_act(float* __restrict__ p, const float* __restrict__ bias,
                           int n_nodes, int dout, int do_relu) {
  long long i = (long long)blockIdx.x * blockDim.x + threadIdx.x;
  if (i >= (long long)n_nodes * dout) return;
  float v = p[i] + bias[(int)(i % dout)];
  p[i] = do_relu ? fmaxf(v, 0.0f) : v;
}

__global__ void k_mlp1(const float* __restrict__ h, const float* __restrict__ W,
                       const float* __restrict__ b, float* __restrict__ t1,
                       int n_nodes) {
  int gid = blockIdx.x * blockDim.x + threadIdx.x;
  if (gid >= n_nodes * 32) return;
  int n = gid >> 5, j = gid & 31;
  const float* r = h + (size_t)n * 64;
  float s = b[j];
#pragma unroll 8
  for (int k = 0; k < 64; ++k) s += r[k] * W[k * 32 + j];
  t1[(size_t)n * 32 + j] = fmaxf(s, 0.0f);
}

__global__ void k_mlp2(const float* __restrict__ t1, const float* __restrict__ W,
                       const float* __restrict__ b, float* __restrict__ out,
                       int n_nodes) {
  int n = blockIdx.x * blockDim.x + threadIdx.x;
  if (n >= n_nodes) return;
  const float* r = t1 + (size_t)n * 32;
  float s = b[0];
#pragma unroll
  for (int k = 0; k < 32; ++k) s += r[k] * W[k];
  out[n] = s;
}

// -------------------- host launcher --------------------

extern "C" void kernel_launch(void* const* d_in, const int* in_sizes, int n_in,
                              void* d_out, int out_size, void* d_ws, size_t ws_size,
                              hipStream_t stream) {
  const float* x  = (const float*)d_in[0];
  const int*   ei = (const int*)d_in[1];
  const float* Wp = (const float*)d_in[2];
  const float* bp = (const float*)d_in[3];
  const float* Wl[4]  = {(const float*)d_in[4],  (const float*)d_in[8],
                         (const float*)d_in[12], (const float*)d_in[16]};
  const float* ASl[4] = {(const float*)d_in[5],  (const float*)d_in[9],
                         (const float*)d_in[13], (const float*)d_in[17]};
  const float* ADl[4] = {(const float*)d_in[6],  (const float*)d_in[10],
                         (const float*)d_in[14], (const float*)d_in[18]};
  const float* Bl[4]  = {(const float*)d_in[7],  (const float*)d_in[11],
                         (const float*)d_in[15], (const float*)d_in[19]};
  const float* Wo1 = (const float*)d_in[20];
  const float* bo1 = (const float*)d_in[21];
  const float* Wo2 = (const float*)d_in[22];
  const float* bo2 = (const float*)d_in[23];

  const int n_nodes = in_sizes[0] / 16;
  const int n_edges = in_sizes[1] / 2;
  const int* srcs = ei;
  const int* dsts = ei + n_edges;

  const int din_l[4]  = {64, 256, 256, 256};
  const int dout_l[4] = {256, 256, 256, 64};
  const int H_l[4]    = {4, 4, 4, 1};

  // ---- carve workspace ----
  char*  ws  = (char*)d_ws;
  size_t off = 0;
  auto carve = [&](size_t bytes) -> void* {
    void* p = ws + off;
    off += (bytes + 255) & ~(size_t)255;
    return p;
  };
  const size_t nf = (size_t)n_nodes * 256;
  float* buf[3];
  for (int i = 0; i < 3; ++i) buf[i] = (float*)carve(nf * sizeof(float));
  float* asrc  = (float*)carve((size_t)n_nodes * 4 * sizeof(float));
  float* adst  = (float*)carve((size_t)n_nodes * 4 * sizeof(float));
  float* emax  = (float*)carve((size_t)n_nodes * 4 * sizeof(float));
  float* denom = (float*)carve((size_t)n_nodes * 4 * sizeof(float));
  _Float16* wt = (_Float16*)carve((size_t)256 * 256 * sizeof(_Float16));

  const int TB = 256;
  const int m_tiles = (n_nodes + 15) / 16;
  const int etot    = n_edges + n_nodes;

  // ---- input projection: buf[0] = relu(x @ Wp + bp) ----
  k_proj<<<(n_nodes * 64 + TB - 1) / TB, TB, 0, stream>>>(x, Wp, bp, buf[0], n_nodes);

  // ---- GAT layers ----
  int ih = 0;
  for (int l = 0; l < 4; ++l) {
    const int din = din_l[l], dout = dout_l[l], H = H_l[l];
    const int ip = (ih + 1) % 3, io = (ih + 2) % 3;

    k_convert_wt<<<(din * dout + TB - 1) / TB, TB, 0, stream>>>(Wl[l], wt, din, dout);

    dim3 gg(m_tiles, (dout + 127) / 128);
    if (din == 64)
      k_gemm_wmma<64><<<gg, 128, 0, stream>>>(buf[ih], wt, buf[ip], n_nodes, dout);
    else
      k_gemm_wmma<256><<<gg, 128, 0, stream>>>(buf[ih], wt, buf[ip], n_nodes, dout);

    k_alpha_init<<<(n_nodes * H + TB - 1) / TB, TB, 0, stream>>>(
        buf[ip], ASl[l], ADl[l], asrc, adst, emax, denom, n_nodes, H);

    long long nout = (long long)n_nodes * dout;
    k_zero<<<(int)((nout + TB - 1) / TB), TB, 0, stream>>>(buf[io], nout);

    k_edge_max<<<(etot + TB - 1) / TB, TB, 0, stream>>>(
        srcs, dsts, n_edges, n_nodes, asrc, adst, emax, H);
    k_edge_sum<<<(etot + TB - 1) / TB, TB, 0, stream>>>(
        srcs, dsts, n_edges, n_nodes, asrc, adst, emax, denom, H);

    long long tmsg = (long long)etot * 64;
    k_edge_msg<<<(int)((tmsg + TB - 1) / TB), TB, 0, stream>>>(
        srcs, dsts, n_edges, n_nodes, asrc, adst, emax, denom, buf[ip], buf[io], H);

    k_bias_act<<<(int)((nout + TB - 1) / TB), TB, 0, stream>>>(
        buf[io], Bl[l], n_nodes, dout, l < 3 ? 1 : 0);

    ih = io;
  }

  // ---- output MLP: t1 = relu(h @ Wo1 + bo1); out = t1 @ Wo2 + bo2 ----
  float* t1 = buf[(ih + 1) % 3];
  k_mlp1<<<(n_nodes * 32 + TB - 1) / TB, TB, 0, stream>>>(buf[ih], Wo1, bo1, t1, n_nodes);
  k_mlp2<<<(n_nodes + TB - 1) / TB, TB, 0, stream>>>(t1, Wo2, bo2, (float*)d_out, n_nodes);
}
